// ConvLayer_1709396984468
// MI455X (gfx1250) — matrix operand order repaired and checked by reading, not compiled, not run on hardware
//
#include <hip/hip_runtime.h>
#include <hip/hip_bf16.h>
#include <math.h>

typedef __attribute__((ext_vector_type(16))) __bf16          v16bf;
typedef __attribute__((ext_vector_type(16))) unsigned short  v16u;
typedef __attribute__((ext_vector_type(8)))  float           v8f;

#define NATOM   100000
#define MNBR    12
#define ATOMF   64
#define NBRF    41
#define KDIM    169          // 2*64 + 41
#define TWOF    128
#define NROWS   (NATOM * MNBR)   // 1,200,000
#define BROWS   128              // rows per block
#define NKT     6                // ceil(169/32) K-tiles of 32
#define LEAK    0.01f
#define EPS     1e-5f

static __device__ __forceinline__ v16bf as_bf16x16(v16u u) {
    return __builtin_bit_cast(v16bf, u);
}

static __device__ __forceinline__ void split_bf16(float x, unsigned short& hs, unsigned short& ls) {
    unsigned ux = __float_as_uint(x);
    unsigned hb = ux & 0xffff0000u;                 // truncate to bf16
    float lof = x - __uint_as_float(hb);            // residual
    hs = (unsigned short)(hb >> 16);
    ls = (unsigned short)(__float_as_uint(lof) >> 16);
}

// ---------------------------------------------------------------------------
// Kernel 0: zero the stats scratch (bn1 sum/sq: 256 floats, bn2 sum/sq: 128)
// ---------------------------------------------------------------------------
__global__ void init_stats_kernel(float* bn1_stats, float* bn2_stats) {
    int t = threadIdx.x;
    if (t < 256) bn1_stats[t] = 0.f;
    if (t < 128) bn2_stats[t] = 0.f;
}

// ---------------------------------------------------------------------------
// Kernel 1: fused gather+concat+GEMM (split-bf16 WMMA) + bias + bn1 stats
//   gated[row, o] = sum_k X[row,k] * W[o,k] + b[o]
//   X[row, 0:64]   = atom[n],  X[row,64:128] = atom[idx[n,m]],
//   X[row,128:169] = nbr[n,m]
// 256 threads = 8 waves; each wave owns 16 rows, all 128 output columns.
// ---------------------------------------------------------------------------
__global__ __launch_bounds__(256) void gemm_bn1_kernel(
    const float* __restrict__ atom, const float* __restrict__ nbr,
    const float* __restrict__ W,    const float* __restrict__ bvec,
    const int*   __restrict__ idx,
    float* __restrict__ gated, float* __restrict__ bn1_sum, float* __restrict__ bn1_sq)
{
    __shared__ __align__(16) unsigned short sAhi[8 * 32 * 16];  // [wave][lane][half]
    __shared__ __align__(16) unsigned short sAlo[8 * 32 * 16];
    __shared__ __align__(16) unsigned short sBhi[8 * 32 * 16];  // [ctile][lane][half]
    __shared__ __align__(16) unsigned short sBlo[8 * 32 * 16];
    __shared__ float sChSum[TWOF];
    __shared__ float sChSq[TWOF];

    const int tid  = threadIdx.x;
    const int lane = tid & 31;
    const int wave = tid >> 5;
    const int row_block = blockIdx.x * BROWS;

    if (tid < TWOF) { sChSum[tid] = 0.f; sChSq[tid] = 0.f; }

    v8f acc[8];
#pragma unroll
    for (int c = 0; c < 8; ++c)
#pragma unroll
        for (int r = 0; r < 8; ++r) acc[c][r] = 0.f;

    // staging assignment: thread -> (local row = tid>>1, k-half = (tid&1)*16)
    const int srow  = tid >> 1;
    const int khalf = (tid & 1) << 4;
    const int grow  = row_block + srow;
    const int n_    = grow / MNBR;
    const int j_    = idx[grow];
    const float* selfp = atom + (size_t)n_ * ATOMF;
    const float* nbrp  = atom + (size_t)j_ * ATOMF;
    const float* bondp = nbr  + (size_t)grow * NBRF;

    const int a_lanebase = (srow & 15);
    const int a_wavebase = (srow >> 4) * 32 * 16;

    const int b_o      = tid >> 1;                       // output channel 0..127
    const int b_ct     = b_o >> 4;
    const int b_lane   = (b_o & 15) + ((tid & 1) << 4);
    const int b_base   = (b_ct * 32 + b_lane) * 16;
    const float* wp    = W + (size_t)b_o * KDIM;

    for (int kt = 0; kt < NKT; ++kt) {
        __syncthreads();
        const int k0 = kt * 32;
        // ---- stage A (X tile, hi/lo bf16, fragment order) ----
#pragma unroll
        for (int i = 0; i < 16; ++i) {
            const int kk = khalf + i;          // 0..31 within tile
            const int k  = k0 + kk;            // absolute feature index
            float x;
            if      (k < ATOMF)      x = selfp[k];
            else if (k < 2 * ATOMF)  x = nbrp[k - ATOMF];
            else if (k < KDIM)       x = bondp[k - 2 * ATOMF];
            else                     x = 0.f;
            unsigned short hs, ls;
            split_bf16(x, hs, ls);
            const int alane = a_lanebase + ((kk >> 4) << 4);
            const int h     = kk & 15;
            const int o     = a_wavebase + alane * 16 + h;
            sAhi[o] = hs; sAlo[o] = ls;
        }
        // ---- stage B (W tile, hi/lo bf16, fragment order) ----
#pragma unroll
        for (int i = 0; i < 16; ++i) {
            const int k = k0 + khalf + i;
            float x = (k < KDIM) ? wp[k] : 0.f;
            unsigned short hs, ls;
            split_bf16(x, hs, ls);
            sBhi[b_base + i] = hs; sBlo[b_base + i] = ls;
        }
        __syncthreads();
        // ---- WMMA compute: acc += Ahi*Bhi + Ahi*Blo + Alo*Bhi ----
        const int abase = (wave * 32 + lane) * 16;
        v16bf ahi = as_bf16x16(*(const v16u*)&sAhi[abase]);
        v16bf alo = as_bf16x16(*(const v16u*)&sAlo[abase]);
#pragma unroll
        for (int ct = 0; ct < 8; ++ct) {
            const int bbase = (ct * 32 + lane) * 16;
            v16bf bhi = as_bf16x16(*(const v16u*)&sBhi[bbase]);
            v16bf blo = as_bf16x16(*(const v16u*)&sBlo[bbase]);
            acc[ct] = __builtin_amdgcn_wmma_f32_16x16x32_bf16(
                false, ahi, false, bhi, (short)0, acc[ct], false, false);
            acc[ct] = __builtin_amdgcn_wmma_f32_16x16x32_bf16(
                false, ahi, false, blo, (short)0, acc[ct], false, false);
            acc[ct] = __builtin_amdgcn_wmma_f32_16x16x32_bf16(
                false, alo, false, bhi, (short)0, acc[ct], false, false);
        }
    }

    // ---- epilogue: bias, store gated, per-channel sum/sumsq ----
    // D layout: VGPR r -> M = r + 8*(lane>=16), N = lane&15
    const int col_n = lane & 15;
    const int wrow  = row_block + wave * 16 + ((lane >> 4) << 3);
#pragma unroll
    for (int ct = 0; ct < 8; ++ct) {
        const int col = ct * 16 + col_n;
        const float bb = bvec[col];
        float s = 0.f, q = 0.f;
#pragma unroll
        for (int r = 0; r < 8; ++r) {
            float v = acc[ct][r] + bb;
            gated[(size_t)(wrow + r) * TWOF + col] = v;
            s += v; q += v * v;
        }
        atomicAdd(&sChSum[col], s);
        atomicAdd(&sChSq[col], q);
    }
    __syncthreads();
    if (tid < TWOF) {
        atomicAdd(&bn1_sum[tid], sChSum[tid]);
        atomicAdd(&bn1_sq[tid],  sChSq[tid]);
    }
}

// ---------------------------------------------------------------------------
// Kernel 2: bn1 finalize -> scale/shift (128 channels, 1 block)
// ---------------------------------------------------------------------------
__global__ void bn1_finalize_kernel(const float* __restrict__ sum, const float* __restrict__ sq,
                                    const float* __restrict__ gamma, const float* __restrict__ beta,
                                    float* __restrict__ scale, float* __restrict__ shift)
{
    int c = threadIdx.x;
    if (c >= TWOF) return;
    const float inv = 1.f / (float)NROWS;
    float mean = sum[c] * inv;
    float var  = sq[c] * inv - mean * mean;
    float sc   = rsqrtf(var + EPS) * gamma[c];
    scale[c] = sc;
    shift[c] = beta[c] - mean * sc;
}

// ---------------------------------------------------------------------------
// Kernel 3: apply bn1, sigmoid-gate * leaky-core, reduce over M
//   256 threads cover 4 atoms x 64 channels
// ---------------------------------------------------------------------------
__global__ __launch_bounds__(256) void gate_reduce_kernel(
    const float* __restrict__ gated, const float* __restrict__ scale, const float* __restrict__ shift,
    float* __restrict__ nsum)
{
    const int tid = threadIdx.x;
    const int f  = tid & 63;
    const int nn = blockIdx.x * 4 + (tid >> 6);
    const float sc1 = scale[f],      sh1 = shift[f];
    const float sc2 = scale[64 + f], sh2 = shift[64 + f];
    const size_t base = (size_t)nn * MNBR * TWOF;
    float acc = 0.f;
#pragma unroll
    for (int m = 0; m < MNBR; ++m) {
        float g1 = gated[base + m * TWOF + f];
        float g2 = gated[base + m * TWOF + 64 + f];
        float a1 = g1 * sc1 + sh1;
        float a2 = g2 * sc2 + sh2;
        float sig  = 1.f / (1.f + __expf(-a1));
        float core = (a2 >= 0.f) ? a2 : LEAK * a2;
        acc += sig * core;
    }
    nsum[(size_t)nn * ATOMF + f] = acc;
}

// ---------------------------------------------------------------------------
// Kernel 4: bn2 stats over nbr_sumed (N x 64)
// ---------------------------------------------------------------------------
__global__ __launch_bounds__(256) void bn2_stats_kernel(
    const float* __restrict__ nsum, float* __restrict__ sum, float* __restrict__ sq)
{
    __shared__ float ls[ATOMF], lq[ATOMF];
    const int tid = threadIdx.x;
    const int f = tid & 63;
    if (tid < ATOMF) { ls[tid] = 0.f; lq[tid] = 0.f; }
    __syncthreads();
    float s = 0.f, q = 0.f;
    for (int nn = blockIdx.x * 4 + (tid >> 6); nn < NATOM; nn += gridDim.x * 4) {
        float x = nsum[(size_t)nn * ATOMF + f];
        s += x; q += x * x;
    }
    atomicAdd(&ls[f], s);
    atomicAdd(&lq[f], q);
    __syncthreads();
    if (tid < ATOMF) {
        atomicAdd(&sum[tid], ls[tid]);
        atomicAdd(&sq[tid],  lq[tid]);
    }
}

// ---------------------------------------------------------------------------
// Kernel 5: bn2 finalize (64 channels)
// ---------------------------------------------------------------------------
__global__ void bn2_finalize_kernel(const float* __restrict__ sum, const float* __restrict__ sq,
                                    const float* __restrict__ gamma, const float* __restrict__ beta,
                                    float* __restrict__ scale, float* __restrict__ shift)
{
    int c = threadIdx.x;
    if (c >= ATOMF) return;
    const float inv = 1.f / (float)NATOM;
    float mean = sum[c] * inv;
    float var  = sq[c] * inv - mean * mean;
    float sc   = rsqrtf(var + EPS) * gamma[c];
    scale[c] = sc;
    shift[c] = beta[c] - mean * sc;
}

// ---------------------------------------------------------------------------
// Kernel 6: residual + leaky
// ---------------------------------------------------------------------------
__global__ __launch_bounds__(256) void residual_kernel(
    const float* __restrict__ atom, const float* __restrict__ nsum,
    const float* __restrict__ scale, const float* __restrict__ shift,
    float* __restrict__ out)
{
    const size_t i = (size_t)blockIdx.x * 256 + threadIdx.x;
    const int f = (int)(i & 63);
    float v = atom[i] + nsum[i] * scale[f] + shift[f];
    out[i] = (v >= 0.f) ? v : LEAK * v;
}

// ---------------------------------------------------------------------------
extern "C" void kernel_launch(void* const* d_in, const int* in_sizes, int n_in,
                              void* d_out, int out_size, void* d_ws, size_t ws_size,
                              hipStream_t stream)
{
    const float* atom      = (const float*)d_in[0];   // (N, 64)
    const float* nbr       = (const float*)d_in[1];   // (N, 12, 41)
    const float* W         = (const float*)d_in[2];   // (128, 169)
    const float* bvec      = (const float*)d_in[3];   // (128,)
    const float* bn1_gamma = (const float*)d_in[4];
    const float* bn1_beta  = (const float*)d_in[5];
    const float* bn2_gamma = (const float*)d_in[6];
    const float* bn2_beta  = (const float*)d_in[7];
    const int*   idx       = (const int*)d_in[8];     // (N, 12)
    float* out = (float*)d_out;                       // (N, 64)

    // workspace layout (floats)
    float* w = (float*)d_ws;
    float* gated     = w;                                   // 153,600,000
    float* bn1_stats = gated + (size_t)NROWS * TWOF;        // sum[128] | sq[128]
    float* bn1_sum   = bn1_stats;
    float* bn1_sq    = bn1_stats + 128;
    float* bn1_scale = bn1_stats + 256;
    float* bn1_shift = bn1_stats + 384;
    float* nsum      = bn1_stats + 512;                     // 6,400,000
    float* bn2_stats = nsum + (size_t)NATOM * ATOMF;
    float* bn2_sum   = bn2_stats;
    float* bn2_sq    = bn2_stats + 64;
    float* bn2_scale = bn2_stats + 128;
    float* bn2_shift = bn2_stats + 192;

    init_stats_kernel<<<1, 384, 0, stream>>>(bn1_stats, bn2_stats);

    gemm_bn1_kernel<<<NROWS / BROWS, 256, 0, stream>>>(
        atom, nbr, W, bvec, idx, gated, bn1_sum, bn1_sq);

    bn1_finalize_kernel<<<1, 128, 0, stream>>>(
        bn1_sum, bn1_sq, bn1_gamma, bn1_beta, bn1_scale, bn1_shift);

    gate_reduce_kernel<<<NATOM / 4, 256, 0, stream>>>(
        gated, bn1_scale, bn1_shift, nsum);

    bn2_stats_kernel<<<512, 256, 0, stream>>>(nsum, bn2_sum, bn2_sq);

    bn2_finalize_kernel<<<1, 64, 0, stream>>>(
        bn2_sum, bn2_sq, bn2_gamma, bn2_beta, bn2_scale, bn2_shift);

    residual_kernel<<<(NATOM * ATOMF) / 256, 256, 0, stream>>>(
        atom, nsum, bn2_scale, bn2_shift, out);
}
